// Network_2654289789634
// MI455X (gfx1250) — matrix operand order repaired and checked by reading, not compiled, not run on hardware
//
#include <hip/hip_runtime.h>
#include <hip/hip_bf16.h>
#include <cstdint>
#include <cstddef>

#define WIDTH    4096
#define DEPTH    512
#define N_HIST   20
#define N_SCALES 64

// step-kernel blocking
#define KC        128                 // K elements staged per pipeline stage
#define NSTAGE    (WIDTH / KC)        // 32
#define PITCH     (KC + 8)            // +16B row pad: 4-bank rotation, keeps 16B align
#define LDS_BUF   (64 * PITCH)        // elems per (matrix, buffer)
#define SMEM_SH   (4 * LDS_BUF * 2)   // bytes: {A,B} x {buf0,buf1} bf16

typedef __attribute__((ext_vector_type(16))) __bf16 v16bf;
typedef __attribute__((ext_vector_type(8)))  float  v8f;

union BF16x16 { v16bf v; uint4 q[2]; };
union BF16x8  { unsigned short h[8]; uint4 q; };

__device__ __forceinline__ unsigned short f2bf_bits(float f) {
    union { float f; uint32_t u; } in; in.f = f;
    uint32_t u = in.u;
    uint32_t r = u + 0x7FFFu + ((u >> 16) & 1u);   // round-to-nearest-even
    return (unsigned short)(r >> 16);
}

__device__ __forceinline__ uint4 pack8(float4 a, float4 b) {
    BF16x8 o;
    o.h[0] = f2bf_bits(a.x); o.h[1] = f2bf_bits(a.y);
    o.h[2] = f2bf_bits(a.z); o.h[3] = f2bf_bits(a.w);
    o.h[4] = f2bf_bits(b.x); o.h[5] = f2bf_bits(b.y);
    o.h[6] = f2bf_bits(b.z); o.h[7] = f2bf_bits(b.w);
    return o.q;
}

// ---------------------------------------------------------------------------
// 1) W (fp32, 4096x4096) -> bf16 workspace.
// ---------------------------------------------------------------------------
__global__ __launch_bounds__(256)
void cvt_f32_bf16_kernel(const float* __restrict__ src,
                         unsigned short* __restrict__ dst, int n_vec8) {
    int i = blockIdx.x * blockDim.x + threadIdx.x;
    if (i >= n_vec8) return;
    const float4* s = (const float4*)src + 2 * (size_t)i;
    ((uint4*)dst)[i] = pack8(s[0], s[1]);
}

// ---------------------------------------------------------------------------
// 2) Initial state: 64 identical rows of bf16(x).
// ---------------------------------------------------------------------------
__global__ __launch_bounds__(256)
void init_state_kernel(const float* __restrict__ x, unsigned short* __restrict__ st) {
    int j8 = blockIdx.x * blockDim.x + threadIdx.x;   // 0..511
    if (j8 >= WIDTH / 8) return;
    const float4* s = (const float4*)x + 2 * (size_t)j8;
    uint4 p = pack8(s[0], s[1]);
    uint4* d = (uint4*)st + j8;
#pragma unroll 4
    for (int m = 0; m < N_SCALES; ++m) d[(size_t)m * (WIDTH / 8)] = p;
}

// ---------------------------------------------------------------------------
// 3) Prologue GEMM (one-time): bmat[m,n] = sum_k biases[m,k] * W_bias[n,k]
//    fp32 -> bf16 in registers, fp32 WMMA accumulation. One wave per 32x16.
// ---------------------------------------------------------------------------
__global__ __launch_bounds__(32)
void bias_gemm_kernel(const float* __restrict__ A,   // biases 512x4096
                      const float* __restrict__ B,   // W_bias 4096x4096
                      float* __restrict__ bmat) {    // 512x4096
    const int lane = threadIdx.x;
    const int hi   = lane >> 4;
    const int lidx = lane & 15;
    const int m0   = blockIdx.x * 32;
    const int n0   = blockIdx.y * 16;

    const float* arow0 = A + (size_t)(m0 + lidx) * WIDTH;
    const float* arow1 = arow0 + (size_t)16 * WIDTH;
    const float* brow  = B + (size_t)(n0 + lidx) * WIDTH + hi * 8;
    const int akoff = hi * 8;

    v8f acc0 = {}; v8f acc1 = {};

    for (int k0 = 0; k0 < WIDTH; k0 += 32) {
        const float4* pa0 = (const float4*)(arow0 + k0 + akoff);
        const float4* pa1 = (const float4*)(arow1 + k0 + akoff);
        const float4* pb  = (const float4*)(brow  + k0);
        BF16x16 a0, a1, b;
        a0.q[0] = pack8(pa0[0], pa0[1]);
        a0.q[1] = pack8(pa0[4], pa0[5]);     // +16 floats
        a1.q[0] = pack8(pa1[0], pa1[1]);
        a1.q[1] = pack8(pa1[4], pa1[5]);
        b.q[0]  = pack8(pb[0],  pb[1]);
        b.q[1]  = pack8(pb[4],  pb[5]);
        acc0 = __builtin_amdgcn_wmma_f32_16x16x32_bf16(false, a0.v, false, b.v,
                                                       (short)0, acc0, false, false);
        acc1 = __builtin_amdgcn_wmma_f32_16x16x32_bf16(false, a1.v, false, b.v,
                                                       (short)0, acc1, false, false);
    }

    const int n = n0 + lidx;
#pragma unroll
    for (int v = 0; v < 8; ++v) {
        int ma = m0 + hi * 8 + v;
        int mb = ma + 16;
        bmat[(size_t)ma * WIDTH + n] = acc0[v];
        bmat[(size_t)mb * WIDTH + n] = acc1[v];
    }
}

// ---------------------------------------------------------------------------
// 4) One recurrence step: next = erf(scale_m * (curr @ W^T) + b_s) / 64.
//    Block = 16 waves computing a 64x64 tile; A and B K-slices double-buffered
//    through LDS (one s_barrier per stage, global loads overlap WMMA).
//    Per-step global traffic: curr 32MB (512KB read by each of 64 blocks) + W 32MB (each row by one block).
// ---------------------------------------------------------------------------
__global__ __launch_bounds__(512)
void step_kernel(const unsigned short* __restrict__ curr,   // 64x4096 bf16
                 const unsigned short* __restrict__ Wbf,    // 4096x4096 bf16
                 const float* __restrict__ bmat,            // 512x4096 fp32
                 const float* __restrict__ scales,          // 64 fp32
                 unsigned short* __restrict__ next,         // 64x4096 bf16
                 float* __restrict__ out,                   // 64x20x4096 fp32
                 int stp) {
    extern __shared__ unsigned short smem[];
    unsigned short* As = smem;                 // [2][64][PITCH]
    unsigned short* Bs = smem + 2 * LDS_BUF;   // [2][64][PITCH]

    const int tid  = threadIdx.x;
    const int lane = tid & 31;
    const int wid  = tid >> 5;                 // 0..15
    const int hi   = lane >> 4;
    const int lidx = lane & 15;
    const int m0   = (wid & 3) * 16;           // wave's M tile
    const int n0l  = (wid >> 2) * 16;          // wave's N tile (block-local)
    const int nblk = blockIdx.x * 64;

    // --- staging assignment: thread t moves 2 A-chunks + 2 B-chunks (16B each)
    const int srow = tid >> 4;                 // 0..31
    const int scol = (tid & 15) * 8;           // 0..120
    const unsigned short* Ag0 = curr + (size_t)srow * WIDTH + scol;
    const unsigned short* Ag1 = Ag0 + (size_t)32 * WIDTH;
    const unsigned short* Bg0 = Wbf + (size_t)(nblk + srow) * WIDTH + scol;
    const unsigned short* Bg1 = Bg0 + (size_t)32 * WIDTH;
    unsigned short* Asw0 = As + srow * PITCH + scol;
    unsigned short* Asw1 = Asw0 + 32 * PITCH;
    unsigned short* Bsw0 = Bs + srow * PITCH + scol;
    unsigned short* Bsw1 = Bsw0 + 32 * PITCH;

    // prologue: stage 0 into buffer 0
    {
        uint4 a0 = *(const uint4*)Ag0;
        uint4 a1 = *(const uint4*)Ag1;
        uint4 b0 = *(const uint4*)Bg0;
        uint4 b1 = *(const uint4*)Bg1;
        *(uint4*)Asw0 = a0;  *(uint4*)Asw1 = a1;
        *(uint4*)Bsw0 = b0;  *(uint4*)Bsw1 = b1;
    }
    __syncthreads();

    v8f acc = {};
    const unsigned short* Ar = As + (m0 + lidx) * PITCH;
    const unsigned short* Br = Bs + (n0l + lidx) * PITCH;
    const int akoff = hi * 8;
    const int bkoff = hi * 8;

    for (int s = 0; s < NSTAGE; ++s) {
        // issue global loads for stage s+1 (wraps harmlessly on last iter)
        const int kn = ((s + 1) & (NSTAGE - 1)) * KC;
        uint4 a0 = *(const uint4*)(Ag0 + kn);
        uint4 a1 = *(const uint4*)(Ag1 + kn);
        uint4 b0 = *(const uint4*)(Bg0 + kn);
        uint4 b1 = *(const uint4*)(Bg1 + kn);
        __builtin_prefetch(Bg0 + ((s + 2) & (NSTAGE - 1)) * KC, 0, 1);

        // compute stage s out of buf[s&1]
        const int bufo = (s & 1) * LDS_BUF;
#pragma unroll
        for (int kk = 0; kk < KC; kk += 32) {
            BF16x16 a, b;
            a.q[0] = *(const uint4*)(Ar + bufo + kk + akoff);
            a.q[1] = *(const uint4*)(Ar + bufo + kk + 16 + akoff);
            b.q[0] = *(const uint4*)(Br + bufo + kk + bkoff);
            b.q[1] = *(const uint4*)(Br + bufo + kk + bkoff + 16);
            acc = __builtin_amdgcn_wmma_f32_16x16x32_bf16(false, a.v, false, b.v,
                                                          (short)0, acc, false, false);
        }

        // commit stage s+1 into the other buffer (held stage s-1: all waves
        // finished it before the previous barrier), then one barrier.
        const int bufn = ((s + 1) & 1) * LDS_BUF;
        *(uint4*)(Asw0 + bufn) = a0;  *(uint4*)(Asw1 + bufn) = a1;
        *(uint4*)(Bsw0 + bufn) = b0;  *(uint4*)(Bsw1 + bufn) = b1;
        __syncthreads();
    }

    // epilogue: scale, bias, erf, 1/sqrt(W); write bf16 state + fp32 history
    const int n = nblk + n0l + lidx;
    const float bias = bmat[(size_t)stp * WIDTH + n];
    const float inv_sqrt_w = 0.015625f;            // 1/sqrt(4096)
    const bool  hist = (stp >= DEPTH - N_HIST);
    float* outp = out + (size_t)(stp - (DEPTH - N_HIST)) * WIDTH + n;

#pragma unroll
    for (int v = 0; v < 8; ++v) {
        int m = m0 + hi * 8 + v;                   // D layout: VGPR v -> row v (+8 hi half)
        float val = erff(scales[m] * acc[v] + bias) * inv_sqrt_w;
        next[(size_t)m * WIDTH + n] = f2bf_bits(val);
        if (hist) outp[(size_t)m * (N_HIST * WIDTH)] = val;
    }
}

// ---------------------------------------------------------------------------
// Host-side orchestration (graph-capture safe: only kernel launches on stream)
// ---------------------------------------------------------------------------
extern "C" void kernel_launch(void* const* d_in, const int* in_sizes, int n_in,
                              void* d_out, int out_size, void* d_ws, size_t ws_size,
                              hipStream_t stream) {
    (void)in_sizes; (void)n_in; (void)out_size; (void)ws_size;

    const float* x       = (const float*)d_in[0];   // (4096)
    const float* biases  = (const float*)d_in[1];   // (512, 4096)
    const float* W       = (const float*)d_in[2];   // (4096, 4096)
    const float* W_bias  = (const float*)d_in[3];   // (4096, 4096)
    const float* scales  = (const float*)d_in[4];   // (64)
    float* out = (float*)d_out;                     // (64, 20, 4096)

    char* ws = (char*)d_ws;
    unsigned short* Wbf  = (unsigned short*)ws;                         // 33,554,432 B
    float*          bmat = (float*)(ws + 33554432);                     //  8,388,608 B
    unsigned short* stA  = (unsigned short*)(ws + 33554432 + 8388608);  //    524,288 B
    unsigned short* stB  = stA + (size_t)N_SCALES * WIDTH;              //    524,288 B

    // 1) W -> bf16
    {
        int nv8 = (WIDTH * WIDTH) / 8;
        cvt_f32_bf16_kernel<<<(nv8 + 255) / 256, 256, 0, stream>>>(W, Wbf, nv8);
    }
    // 2) initial state
    init_state_kernel<<<2, 256, 0, stream>>>(x, stA);
    // 3) prologue GEMM: bmat = biases @ W_bias^T
    bias_gemm_kernel<<<dim3(512 / 32, WIDTH / 16), 32, 0, stream>>>(biases, W_bias, bmat);
    // 4) 512 sequential recurrence steps, ping-pong state buffers
    for (int s = 0; s < DEPTH; ++s) {
        const unsigned short* cur = (s & 1) ? stB : stA;
        unsigned short*       nxt = (s & 1) ? stA : stB;
        step_kernel<<<WIDTH / 64, 512, SMEM_SH, stream>>>(
            cur, Wbf, bmat, scales, nxt, out, s);
    }
}
